// Depth_Synergized_Decoupling_Mamba_31997506355586
// MI455X (gfx1250) — compile-verified, matrix operand order checked by reading, not run on hardware
//
#include <hip/hip_runtime.h>
#include <hip/hip_bf16.h>

#define BB   2
#define CC   192
#define KK   2
#define DS   16
#define DTR  6
#define HH   64
#define WW   64
#define LL   4096
#define NPROJ 38            // DTR + 2*DS
#define SCAN_TILE 64
#define NCHUNK (LL / SCAN_TILE)

typedef __attribute__((ext_vector_type(16))) _Float16     v16h;
typedef __attribute__((ext_vector_type(8)))  float        v8f;
typedef __attribute__((ext_vector_type(4)))  unsigned int u32x4;
typedef __attribute__((ext_vector_type(8)))  int          i32x8;
typedef __attribute__((ext_vector_type(4)))  int          i32x4;

// ===========================================================================
// WMMA fragment helpers (CDNA5 ISA 7.12.2 layouts, wave32)
// A 16x32 f16: lane (m = lane&15, hi = lane>>4); half e<8 -> k = hi*8+e,
//              e>=8 -> k = 16 + hi*8 + (e-8). Both runs are CONTIGUOUS in k,
//              so the A fragment is two 8-float spans of row m: load directly
//              from global with float4s (global_load_b128), no LDS needed.
// B 32x16 f16: mirrored with n = lane&15. Staged into LDS in fragment-major
//              order so the fragment load is one 32B vector (ds_load_b128 x2).
// ===========================================================================
__device__ __forceinline__ v16h load_a_row(const float* __restrict__ row, int hi) {
  const float4* q0 = (const float4*)(row + hi * 8);
  const float4* q1 = (const float4*)(row + 16 + hi * 8);
  float4 f0 = q0[0], f1 = q0[1], f2 = q1[0], f3 = q1[1];
  v16h a;
  a[0]  = (_Float16)f0.x; a[1]  = (_Float16)f0.y; a[2]  = (_Float16)f0.z; a[3]  = (_Float16)f0.w;
  a[4]  = (_Float16)f1.x; a[5]  = (_Float16)f1.y; a[6]  = (_Float16)f1.z; a[7]  = (_Float16)f1.w;
  a[8]  = (_Float16)f2.x; a[9]  = (_Float16)f2.y; a[10] = (_Float16)f2.z; a[11] = (_Float16)f2.w;
  a[12] = (_Float16)f3.x; a[13] = (_Float16)f3.y; a[14] = (_Float16)f3.z; a[15] = (_Float16)f3.w;
  return a;
}

// Stage one B-tile row (k = kk, 16 contiguous floats) into fragment-major LDS.
__device__ __forceinline__ void stage_b_row(_Float16* sb, int kk,
                                            const float* __restrict__ src) {
  int hi = (kk < 16) ? (kk >> 3) : ((kk - 16) >> 3);
  int e  = (kk < 16) ? (kk & 7)  : (8 + ((kk - 16) & 7));
  const float4* q = (const float4*)src;
  float4 f0 = q[0], f1 = q[1], f2 = q[2], f3 = q[3];
  float v[16] = {f0.x, f0.y, f0.z, f0.w, f1.x, f1.y, f1.z, f1.w,
                 f2.x, f2.y, f2.z, f2.w, f3.x, f3.y, f3.z, f3.w};
#pragma unroll
  for (int n = 0; n < 16; ++n)
    sb[(((hi << 4) | n) << 4) + e] = (_Float16)v[n];
}

// Gathered variant: column n comes from src[spos[n]] (sort-permuted columns).
__device__ __forceinline__ void stage_b_gather(_Float16* sb, int kk,
                                               const float* __restrict__ src,
                                               const int* spos) {
  int hi = (kk < 16) ? (kk >> 3) : ((kk - 16) >> 3);
  int e  = (kk < 16) ? (kk & 7)  : (8 + ((kk - 16) & 7));
#pragma unroll
  for (int n = 0; n < 16; ++n)
    sb[(((hi << 4) | n) << 4) + e] = (_Float16)src[spos[n]];
}

__device__ __forceinline__ v8f wmma_ld(const _Float16* sb, int lane, v16h af, v8f acc) {
  v16h bf = *(const v16h*)(sb + (lane << 4));  // contiguous: 2x ds_load_b128
  return __builtin_amdgcn_wmma_f32_16x16x32_f16(false, af, false, bf,
                                                (short)0, acc, false, false);
}

// ===========================================================================
// 1) Fused 3x3 conv -> sort keys (sigmoid(leaky(.)) is monotone -> raw conv
//    output is a valid sort key; prox half shared between both branches).
// ===========================================================================
__global__ void k_conv_keys(const float* __restrict__ xT, const float* __restrict__ xR,
                            const float* __restrict__ prox, const float* __restrict__ fw,
                            const float* __restrict__ fb, float* __restrict__ key) {
  int idx = blockIdx.x * blockDim.x + threadIdx.x;
  if (idx >= BB * LL) return;
  int b = idx / LL, l = idx % LL;
  int y = l >> 6, x = l & 63;
  float aT = 0.f, aR = 0.f, aP = 0.f;
  for (int c = 0; c < CC; ++c) {
    const float* xTb = xT + ((size_t)b * CC + c) * LL;
    const float* xRb = xR + ((size_t)b * CC + c) * LL;
    const float* pb  = prox + ((size_t)b * CC + c) * LL;
#pragma unroll
    for (int dy = 0; dy < 3; ++dy) {
      int yy = y + dy - 1;
      if (yy < 0 || yy >= HH) continue;
#pragma unroll
      for (int dx = 0; dx < 3; ++dx) {
        int xx = x + dx - 1;
        if (xx < 0 || xx >= WW) continue;
        int p = (yy << 6) + xx;
        aT += xTb[p] * fw[(c * 3 + dy) * 3 + dx];
        aR += xRb[p] * fw[(c * 3 + dy) * 3 + dx];
        aP += pb[p]  * fw[((CC + c) * 3 + dy) * 3 + dx];
      }
    }
  }
  key[(b * 2 + 0) * LL + l] = aT + aP + fb[0];
  key[(b * 2 + 1) * LL + l] = aR + aP + fb[0];
}

// ===========================================================================
// 2) In-LDS bitonic argsort of 4096 (key,idx) pairs + inverse permutation.
//    br==0 descending (argsort(-pm)), br==1 ascending; idx-ascending tiebreak
//    reproduces stable argsort in both directions.
// ===========================================================================
__global__ void k_sort(const float* __restrict__ keys, int* __restrict__ order,
                       int* __restrict__ inv) {
  __shared__ float sk[LL];
  __shared__ int   si[LL];
  int br = blockIdx.x, b = blockIdx.y;
  int tid = threadIdx.x;
  bool globalAsc = (br == 1);
  const float* kb = keys + ((size_t)b * 2 + br) * LL;
  for (int i = tid; i < LL; i += 1024) { sk[i] = kb[i]; si[i] = i; }
  __syncthreads();
  for (int ksz = 2; ksz <= LL; ksz <<= 1) {
    for (int j = ksz >> 1; j > 0; j >>= 1) {
      for (int t = tid; t < LL / 2; t += 1024) {
        int i = ((t & ~(j - 1)) << 1) | (t & (j - 1));
        int p = i | j;
        bool effAsc = ((i & ksz) == 0) ? globalAsc : !globalAsc;
        float a = sk[i], c = sk[p];
        int   ai = si[i], ci = si[p];
        bool doswap = effAsc ? ((a > c) || (a == c && ai > ci))
                             : ((a < c) || (a == c && ai > ci));
        if (doswap) { sk[i] = c; sk[p] = a; si[i] = ci; si[p] = ai; }
      }
      __syncthreads();
    }
  }
  int base = (b * 2 + br) * LL;
  for (int i = tid; i < LL; i += 1024) {
    order[base + i] = si[i];
    inv[base + si[i]] = i;
  }
}

// ===========================================================================
// 3) depth_feat: leaky(W[192x192] @ prox + b) — WMMA, A frags from global
// ===========================================================================
__global__ void k_gemm_feat(const float* __restrict__ W, const float* __restrict__ bias,
                            const float* __restrict__ X, float* __restrict__ Y) {
  __shared__ alignas(32) _Float16 sb[512];
  int lane = threadIdx.x;
  int m = lane & 15, hi = lane >> 4;
  int n0 = blockIdx.x * 16, m0 = blockIdx.y * 16, b = blockIdx.z;
  const float* Xb = X + (size_t)b * CC * LL;
  const float* Wrow = W + (size_t)(m0 + m) * CC;
  float* Yb = Y + (size_t)b * CC * LL;
  v8f acc = {};
#pragma unroll
  for (int k0 = 0; k0 < CC; k0 += 32) {
    v16h af = load_a_row(Wrow + k0, hi);
    stage_b_row(sb, lane, Xb + (size_t)(k0 + lane) * LL + n0);
    __syncthreads();
    acc = wmma_ld(sb, lane, af, acc);
    __syncthreads();
  }
  int n = lane & 15;
#pragma unroll
  for (int r = 0; r < 8; ++r) {
    int mm = m0 + r + 8 * hi;
    float v = acc[r] + bias[mm];
    Yb[(size_t)mm * LL + n0 + n] = v > 0.f ? v : 0.2f * v;
  }
}

// ===========================================================================
// 4) depth_sig: sigmoid(leaky(W[16x192] @ prox + b)) — WMMA, single M tile
// ===========================================================================
__global__ void k_gemm_sig(const float* __restrict__ W, const float* __restrict__ bias,
                           const float* __restrict__ X, float* __restrict__ Y) {
  __shared__ alignas(32) _Float16 sb[512];
  int lane = threadIdx.x;
  int m = lane & 15, hi = lane >> 4;
  int n0 = blockIdx.x * 16, b = blockIdx.z;
  const float* Xb = X + (size_t)b * CC * LL;
  const float* Wrow = W + (size_t)m * CC;
  float* Yb = Y + (size_t)b * DS * LL;
  v8f acc = {};
#pragma unroll
  for (int k0 = 0; k0 < CC; k0 += 32) {
    v16h af = load_a_row(Wrow + k0, hi);
    stage_b_row(sb, lane, Xb + (size_t)(k0 + lane) * LL + n0);
    __syncthreads();
    acc = wmma_ld(sb, lane, af, acc);
    __syncthreads();
  }
  int n = lane & 15;
#pragma unroll
  for (int r = 0; r < 8; ++r) {
    int mm = r + 8 * hi;
    float v = acc[r] + bias[mm];
    v = v > 0.f ? v : 0.2f * v;
    Yb[(size_t)mm * LL + n0 + n] = 1.f / (1.f + __expf(-v));
  }
}

// ===========================================================================
// 5) depth_proj[b,k] = dpw[k][32x192] @ feat[b] — WMMA
// ===========================================================================
__global__ void k_gemm_dproj(const float* __restrict__ dpw, const float* __restrict__ feat,
                             float* __restrict__ dproj) {
  __shared__ alignas(32) _Float16 sb[512];
  int lane = threadIdx.x;
  int m = lane & 15, hi = lane >> 4;
  int n0 = blockIdx.x * 16, m0 = blockIdx.y * 16;
  int bk = blockIdx.z, b = bk >> 1, k = bk & 1;
  const float* Wrow = dpw + (size_t)k * 2 * DS * CC + (size_t)(m0 + m) * CC;
  const float* Xb = feat + (size_t)b * CC * LL;
  float* Yb = dproj + (size_t)bk * 2 * DS * LL;
  v8f acc = {};
#pragma unroll
  for (int k0 = 0; k0 < CC; k0 += 32) {
    v16h af = load_a_row(Wrow + k0, hi);
    stage_b_row(sb, lane, Xb + (size_t)(k0 + lane) * LL + n0);
    __syncthreads();
    acc = wmma_ld(sb, lane, af, acc);
    __syncthreads();
  }
  int n = lane & 15;
#pragma unroll
  for (int r = 0; r < 8; ++r)
    Yb[(size_t)(m0 + r + 8 * hi) * LL + n0 + n] = acc[r];
}

// ===========================================================================
// 6) xpe = x + positional encoding (the [:192] PE slice depends only on the
//    x-coordinate: c<96 -> sin(gx*f_c) else cos(gx*f_{c-96})).
// ===========================================================================
__global__ void k_xpe(const float* __restrict__ x, float* __restrict__ xpe) {
  int idx = blockIdx.x * blockDim.x + threadIdx.x;
  if (idx >= BB * CC * LL) return;
  int l = idx & (LL - 1);
  int c = (idx >> 12) % CC;
  float gx = (float)(l & 63) / 63.f;
  int j = c % 96;
  float f = __powf(10000.f, -(float)j / 96.f);
  float pe = (c < 96) ? __sinf(gx * f) : __cosf(gx * f);
  xpe[idx] = x[idx] + pe;
}

// ===========================================================================
// 7) x_dbl[b,k] = x_proj_w[k][38x192] @ gather(xpe[b], m(k,l)) — WMMA with
//    permutation-gathered B columns; M padded to 48 with zero A rows.
// ===========================================================================
__global__ void k_gemm_xdbl(const float* __restrict__ xpw, const float* __restrict__ xpe,
                            const int* __restrict__ order, int br,
                            float* __restrict__ xdbl) {
  __shared__ alignas(32) _Float16 sb[512];
  __shared__ int spos[16];
  int lane = threadIdx.x;
  int m = lane & 15, hi = lane >> 4;
  int n0 = blockIdx.x * 16, m0 = blockIdx.y * 16;
  int bk = blockIdx.z, b = bk >> 1, k = bk & 1;
  const float* Wk = xpw + (size_t)k * NPROJ * CC;
  const float* xp = xpe + (size_t)b * CC * LL;
  const int* ord = order + ((size_t)(b * 2 + br)) * LL;
  if (lane < 16) {
    int l = n0 + lane;
    spos[lane] = (k == 0) ? ord[l] : ord[LL - 1 - l];
  }
  __syncthreads();
  int row = m0 + m;
  bool valid = row < NPROJ;
  const float* Wrow = Wk + (size_t)(valid ? row : (NPROJ - 1)) * CC;
  v8f acc = {};
#pragma unroll
  for (int k0 = 0; k0 < CC; k0 += 32) {
    v16h af = load_a_row(Wrow + k0, hi);
    if (!valid) {
#pragma unroll
      for (int e = 0; e < 16; ++e) af[e] = (_Float16)0.f;
    }
    stage_b_gather(sb, lane, xp + (size_t)(k0 + lane) * LL, spos);
    __syncthreads();
    acc = wmma_ld(sb, lane, af, acc);
    __syncthreads();
  }
  int n = lane & 15;
  float* Yb = xdbl + (size_t)bk * NPROJ * LL;
#pragma unroll
  for (int r = 0; r < 8; ++r) {
    int mm = m0 + r + 8 * hi;
    if (mm < NPROJ) Yb[(size_t)mm * LL + n0 + n] = acc[r];
  }
}

// ===========================================================================
// 8) Selective scan with TDM double-buffering. x_dbl[bk] is a 2D tensor
//    (NPROJ rows x L cols, row stride L); each chunk is a NPROJ x SCAN_TILE
//    tile DMA'd to LDS by the Tensor Data Mover while the recurrence runs on
//    the other buffer. Wave 0 issues tensor_load_to_lds and fences with
//    s_wait_tensorcnt; the workgroup barrier publishes the buffer.
//    D# per ISA 8.3-8.6: group0 {count=1, lds_addr, global_addr, type=2},
//    group1 {data_size=4B, tensor_dim0=L, tensor_dim1=NPROJ,
//            tile_dim0=SCAN_TILE, tile_dim1=NPROJ, dim0_stride=L}.
//    Toolchain here is the 6-arg builtin: (g0,u32x4)(g1,i32x8)(g2,i32x4)
//    (g3,i32x4)(extra,i32x8)(cpol,i32) — groups 2/3/extra zero for 2D.
// ===========================================================================
__device__ __forceinline__ void tdm_load_tile(unsigned lds_off,
                                              const float* __restrict__ gptr) {
  unsigned long long ga = (unsigned long long)(size_t)gptr;
  u32x4 g0 = {};
  g0[0] = 1u;                                            // count=1 (user D#)
  g0[1] = lds_off;                                       // lds_addr (bytes)
  g0[2] = (unsigned)(ga & 0xFFFFFFFFu);                  // global_addr[31:0]
  g0[3] = (unsigned)((ga >> 32) & 0x1FFFFFFu) | (2u << 30); // [56:32] | type=2
  i32x8 g1 = {};
  g1[0] = (2 << 16);                                     // data_size = 4 bytes
  g1[1] = (int)((LL & 0xFFFF) << 16);                    // tensor_dim0 lo16
  g1[2] = (int)(((LL >> 16) & 0xFFFF) | ((NPROJ & 0xFFFF) << 16)); // d0 hi|d1 lo
  g1[3] = (int)(SCAN_TILE << 16);                        // d1 hi | tile_dim0
  g1[4] = (int)(NPROJ & 0xFFFF);                         // tile_dim1 (tile_dim2=0)
  g1[5] = (int)LL;                                       // tensor_dim0_stride lo32
  g1[6] = 0;
  g1[7] = 0;
  i32x4 z4 = {};
  i32x8 z8 = {};
  __builtin_amdgcn_tensor_load_to_lds(g0, g1, z4, z4, z8, 0);
}

__global__ void k_scan(const float* __restrict__ xdbl, const float* __restrict__ dproj,
                       const float* __restrict__ sig, const float* __restrict__ xpe,
                       const int* __restrict__ order, int br,
                       const float* __restrict__ A_log, const float* __restrict__ dtw,
                       const float* __restrict__ dtb, const float* __restrict__ Dp,
                       float* __restrict__ yout) {
  int bk = blockIdx.x, b = bk >> 1, k = bk & 1;
  int c = threadIdx.x;  // channel 0..191
  __shared__ alignas(16) float lxd[2][NPROJ * SCAN_TILE];  // TDM destination
  __shared__ float sB[DS], sC[DS];
  const float* xd = xdbl + (size_t)bk * NPROJ * LL;
  const float* dp = dproj + (size_t)bk * 2 * DS * LL;
  const float* sg = sig + (size_t)b * DS * LL;
  const float* xp = xpe + (size_t)b * CC * LL;
  const int* ord = order + ((size_t)(b * 2 + br)) * LL;
  float* yo = yout + (size_t)bk * CC * LL;
  bool wave0 = (threadIdx.x >> 5) == 0;
  // LDS byte offsets of the two buffers (flat addr [31:0] is the LDS address)
  unsigned lds0 = (unsigned)(size_t)(void*)&lxd[0][0];
  unsigned lds1 = (unsigned)(size_t)(void*)&lxd[1][0];

  float A[DS], h[DS], w6[DTR];
#pragma unroll
  for (int n = 0; n < DS; ++n) {
    A[n] = -__expf(A_log[(size_t)(k * CC + c) * DS + n]);
    h[n] = 0.f;
  }
#pragma unroll
  for (int r = 0; r < DTR; ++r) w6[r] = dtw[(size_t)(k * CC + c) * DTR + r];
  float bconst = dtb[k * CC + c];
  float Dc = Dp[k * CC + c];

  if (wave0) tdm_load_tile(lds0, xd);            // prime chunk 0

  for (int t = 0; t < NCHUNK; ++t) {
    if (wave0) {
      if (t + 1 < NCHUNK) {
        tdm_load_tile((((t + 1) & 1) ? lds1 : lds0), xd + (size_t)(t + 1) * SCAN_TILE);
        __builtin_amdgcn_s_wait_tensorcnt(1);    // chunk t landed
      } else {
        __builtin_amdgcn_s_wait_tensorcnt(0);
      }
    }
    __syncthreads();                             // publish buffer t
    const float* cur = &lxd[t & 1][0];
    for (int j = 0; j < SCAN_TILE; ++j) {
      int l = t * SCAN_TILE + j;
      int pos = (k == 0) ? ord[l] : ord[LL - 1 - l];
      if (c < DS) {
        float w  = sg[(size_t)c * LL + pos];
        float Bo = cur[(DTR + c) * SCAN_TILE + j];
        float Bd = dp[(size_t)c * LL + pos];
        sB[c] = (1.f - w) * Bo + w * Bd;
      } else if (c < 2 * DS) {
        int n = c - DS;
        float w  = sg[(size_t)n * LL + pos];
        float Co = cur[(DTR + DS + n) * SCAN_TILE + j];
        float Cd = dp[(size_t)(DS + n) * LL + pos];
        sC[n] = (1.f - w) * Co + w * Cd;
      }
      if (l + 1 < LL) {                          // prefetch next gathered column
        int pn = (k == 0) ? ord[l + 1] : ord[LL - 2 - l];
        __builtin_prefetch(&xp[(size_t)c * LL + pn], 0, 1);  // global_prefetch_b8
      }
      __syncthreads();
      float draw = bconst;
#pragma unroll
      for (int r = 0; r < DTR; ++r) draw += w6[r] * cur[r * SCAN_TILE + j];
      float dt = (draw > 20.f) ? draw : log1pf(__expf(draw));  // softplus
      float u  = xp[(size_t)c * LL + pos];
      float du = dt * u;
      float y  = Dc * u;
#pragma unroll
      for (int n = 0; n < DS; ++n) {
        h[n] = h[n] * __expf(dt * A[n]) + du * sB[n];
        y += h[n] * sC[n];
      }
      yo[(size_t)c * LL + l] = y;
      __syncthreads();
    }
  }
}

// ===========================================================================
// 9) Un-permute fwd+bwd, layernorm over 192 channels, write output slab.
// ===========================================================================
__global__ void k_finalize(const float* __restrict__ y, const int* __restrict__ inv, int br,
                           const float* __restrict__ g, const float* __restrict__ beta,
                           float* __restrict__ out) {
  int bp = blockIdx.x;
  int b = bp >> 12, pos = bp & (LL - 1);
  int c = threadIdx.x;  // 0..255, first 192 active
  __shared__ float red[256];
  __shared__ float s_mu, s_rs;
  float v = 0.f;
  if (c < CC) {
    int pf = inv[((size_t)(b * 2 + br)) * LL + pos];
    float yf = y[((size_t)((b * 2 + 0) * CC + c)) * LL + pf];
    float yb = y[((size_t)((b * 2 + 1) * CC + c)) * LL + (LL - 1 - pf)];
    v = yf + yb;
  }
  red[c] = (c < CC) ? v : 0.f;
  __syncthreads();
  for (int s = 128; s > 0; s >>= 1) { if (c < s) red[c] += red[c + s]; __syncthreads(); }
  if (c == 0) s_mu = red[0] / (float)CC;
  __syncthreads();
  float mu = s_mu;
  float d = (c < CC) ? (v - mu) : 0.f;
  red[c] = d * d;
  __syncthreads();
  for (int s = 128; s > 0; s >>= 1) { if (c < s) red[c] += red[c + s]; __syncthreads(); }
  if (c == 0) s_rs = rsqrtf(red[0] / (float)CC + 1e-5f);
  __syncthreads();
  if (c < CC)
    out[(((size_t)br * BB + b) * LL + pos) * CC + c] = (v - mu) * s_rs * g[c] + beta[c];
}

// ===========================================================================
extern "C" void kernel_launch(void* const* d_in, const int* in_sizes, int n_in,
                              void* d_out, int out_size, void* d_ws, size_t ws_size,
                              hipStream_t stream) {
  const float* x_T    = (const float*)d_in[0];
  const float* x_R    = (const float*)d_in[1];
  const float* prox   = (const float*)d_in[2];
  const float* fuse_w = (const float*)d_in[3];
  const float* fuse_b = (const float*)d_in[4];
  const float* xpw    = (const float*)d_in[5];
  const float* dtw    = (const float*)d_in[6];
  const float* dtb    = (const float*)d_in[7];
  const float* AlT    = (const float*)d_in[8];
  const float* AlR    = (const float*)d_in[9];
  const float* D_T    = (const float*)d_in[10];
  const float* D_R    = (const float*)d_in[11];
  const float* dencw  = (const float*)d_in[12];
  const float* dencb  = (const float*)d_in[13];
  const float* dpw    = (const float*)d_in[14];
  const float* sigw   = (const float*)d_in[15];
  const float* sigb_w = (const float*)d_in[16];
  const float* ln_g   = (const float*)d_in[17];
  const float* ln_b   = (const float*)d_in[18];
  float* out = (float*)d_out;

  // Workspace carve-up (all L2-resident; ~29 MiB total, 192 MB L2)
  char* wsp = (char*)d_ws;
  size_t off = 0;
  auto carve = [&](size_t bytes) -> void* {
    void* p = wsp + off;
    off = (off + bytes + 255) & ~(size_t)255;
    return p;
  };
  float* key   = (float*)carve((size_t)BB * 2 * LL * 4);
  int*   order = (int*)  carve((size_t)BB * 2 * LL * 4);
  int*   invp  = (int*)  carve((size_t)BB * 2 * LL * 4);
  float* feat  = (float*)carve((size_t)BB * CC * LL * 4);
  float* sigv  = (float*)carve((size_t)BB * DS * LL * 4);
  float* dproj = (float*)carve((size_t)BB * KK * 2 * DS * LL * 4);
  float* xpe   = (float*)carve((size_t)BB * CC * LL * 4);      // reused per branch
  float* xdbl  = (float*)carve((size_t)BB * KK * NPROJ * LL * 4);
  float* yscan = (float*)carve((size_t)BB * KK * CC * LL * 4); // reused per branch
  (void)ws_size; (void)in_sizes; (void)n_in; (void)out_size;

  // Shared pre-computation
  k_conv_keys<<<dim3((BB * LL + 63) / 64), 64, 0, stream>>>(x_T, x_R, prox, fuse_w, fuse_b, key);
  k_sort<<<dim3(2, BB), 1024, 0, stream>>>(key, order, invp);
  k_gemm_feat<<<dim3(LL / 16, CC / 16, BB), 32, 0, stream>>>(dencw, dencb, prox, feat);
  k_gemm_sig<<<dim3(LL / 16, 1, BB), 32, 0, stream>>>(sigw, sigb_w, prox, sigv);
  k_gemm_dproj<<<dim3(LL / 16, 2, BB * KK), 32, 0, stream>>>(dpw, feat, dproj);

  // Two branches, sequential workspace reuse
  for (int br = 0; br < 2; ++br) {
    const float* xin = (br == 0) ? x_T : x_R;
    const float* Al  = (br == 0) ? AlT : AlR;
    const float* Dpv = (br == 0) ? D_T : D_R;
    k_xpe<<<dim3((BB * CC * LL + 255) / 256), 256, 0, stream>>>(xin, xpe);
    k_gemm_xdbl<<<dim3(LL / 16, 3, BB * KK), 32, 0, stream>>>(xpw, xpe, order, br, xdbl);
    k_scan<<<dim3(BB * KK), CC, 0, stream>>>(xdbl, dproj, sigv, xpe, order, br,
                                             Al, dtw, dtb, Dpv, yscan);
    k_finalize<<<dim3(BB * LL), 256, 0, stream>>>(yscan, invp, br, ln_g, ln_b, out);
  }
}
